// SeedGenerator_Deconv_22110491640188
// MI455X (gfx1250) — compile-verified, matrix operand order checked by reading, not run on hardware
//
#include <hip/hip_runtime.h>

typedef __attribute__((ext_vector_type(16))) _Float16 v16h;
typedef __attribute__((ext_vector_type(8)))  float    v8f;

// Per-lane K index map shared by 16-bit A (16x32) and B (32x16) WMMA operands.
// element e in [0,16): K = e%8 + (e/8)*16 + (lane>=16 ? 8 : 0)
__device__ __forceinline__ int kmap(int e, int half) {
  return (e & 7) + ((e >> 3) << 4) + (half << 3);
}

// ---------------------------------------------------------------------------
// Kernel 1: fold feat-broadcast halves of the concat() GEMMs into per-batch
// bias vectors:  fb[b][h] = sum_c W[h][128+c]*feat[b][c] (+ conv biases)
// ---------------------------------------------------------------------------
__global__ void feat_bias_kernel(
    const float* __restrict__ feat,
    const float* __restrict__ m1w1, const float* __restrict__ m1b1,
    const float* __restrict__ m1b2, const float* __restrict__ m1ws,
    const float* __restrict__ m1bs,
    const float* __restrict__ m3w1, const float* __restrict__ m3b1,
    const float* __restrict__ m3b2, const float* __restrict__ m3ws,
    const float* __restrict__ m3bs,
    float* __restrict__ fb1, float* __restrict__ fbs1,
    float* __restrict__ fb3, float* __restrict__ fbs3) {
  int t = blockIdx.x * 256 + threadIdx.x;
  if (t >= 64 * 128) return;
  int b = t >> 7, h = t & 127;
  const float* f   = feat + b * 512;
  const float* r11 = m1w1 + h * 640 + 128;
  const float* r1s = m1ws + h * 640 + 128;
  const float* r31 = m3w1 + h * 640 + 128;
  const float* r3s = m3ws + h * 640 + 128;
  float a1 = 0.f, as1 = 0.f, a3 = 0.f, as3 = 0.f;
  for (int c = 0; c < 512; ++c) {
    float fv = f[c];
    a1  += r11[c] * fv;  as1 += r1s[c] * fv;
    a3  += r31[c] * fv;  as3 += r3s[c] * fv;
  }
  fb1[t]  = a1  + m1b1[h];
  fbs1[t] = as1 + m1bs[h] + m1b2[h];
  fb3[t]  = a3  + m3b1[h];
  fbs3[t] = as3 + m3bs[h] + m3b2[h];
}

// ---------------------------------------------------------------------------
// Kernel 2: deconv GEMM  x1[b, o*256+k] = sum_c feat[b,c] * ps_w[c, o*256+k]
// M=64 (batch), N=32768, K=512.  One wave = one 16x16 tile.
// ---------------------------------------------------------------------------
__global__ __launch_bounds__(128) void deconv_wmma_kernel(
    const float* __restrict__ feat, const float* __restrict__ psw,
    const float* __restrict__ psb, float* __restrict__ out) {
  const int ntiles = 2048;                       // 32768 / 16
  int tile = blockIdx.x * 4 + (threadIdx.x >> 5);
  if (tile >= 4 * ntiles) return;
  int m0 = (tile / ntiles) << 4;
  int n0 = (tile % ntiles) << 4;
  int lane = threadIdx.x & 31;
  int half = lane >> 4;
  int mr   = lane & 15;
  const float* arow = feat + (m0 + mr) * 512;
  v8f acc = {};
  for (int k0 = 0; k0 < 512; k0 += 32) {
    v16h av, bv;
#pragma unroll
    for (int e = 0; e < 16; ++e) {
      int kk = k0 + kmap(e, half);
      av[e] = (_Float16)arow[kk];
      bv[e] = (_Float16)psw[kk * 32768 + n0 + mr];
    }
    acc = __builtin_amdgcn_wmma_f32_16x16x32_f16(false, av, false, bv,
                                                 (short)0, acc, false, false);
  }
  int n = n0 + mr;
  float bias = psb[n >> 8];                      // o = n / 256
#pragma unroll
  for (int r = 0; r < 8; ++r) {
    int m = m0 + r + (half << 3);
    out[m * 32768 + n] = acc[r] + bias;          // [B,128,256] row-major
  }
}

// ---------------------------------------------------------------------------
// Kernel 3: stage-1 of MLP_Res:  Y = relu(W[M,K] @ X[b][K,256] + bias)
// bias indexed [b*biasBS + m] (biasBS = M for per-batch feat bias, 0 shared).
// ---------------------------------------------------------------------------
__global__ __launch_bounds__(128) void mlp_stage1_wmma_kernel(
    const float* __restrict__ W, int ldw,
    const float* __restrict__ X, int xBS,
    float* __restrict__ Y, int yBS,
    const float* __restrict__ bias, int biasBS,
    int M, int K, int doRelu) {
  int tilesPerBatch = (M >> 4) << 4;             // (M/16) * (256/16)
  int tile = blockIdx.x * 4 + (threadIdx.x >> 5);
  if (tile >= 64 * tilesPerBatch) return;
  int b  = tile / tilesPerBatch;
  int t  = tile - b * tilesPerBatch;
  int m0 = (t >> 4) << 4;
  int n0 = (t & 15) << 4;
  int lane = threadIdx.x & 31, half = lane >> 4, mr = lane & 15;
  const float* Xb = X + b * xBS;
  v8f acc = {};
  for (int k0 = 0; k0 < K; k0 += 32) {
    v16h av, bv;
#pragma unroll
    for (int e = 0; e < 16; ++e) {
      int kk = k0 + kmap(e, half);
      av[e] = (_Float16)W[(m0 + mr) * ldw + kk];
      bv[e] = (_Float16)Xb[kk * 256 + n0 + mr];
    }
    acc = __builtin_amdgcn_wmma_f32_16x16x32_f16(false, av, false, bv,
                                                 (short)0, acc, false, false);
  }
  int n = n0 + mr;
#pragma unroll
  for (int r = 0; r < 8; ++r) {
    int m = m0 + r + (half << 3);
    float v = acc[r] + bias[b * biasBS + m];
    if (doRelu) v = fmaxf(v, 0.f);
    Y[b * yBS + m * 256 + n] = v;
  }
}

// ---------------------------------------------------------------------------
// Kernel 4: stage-2 of MLP_Res: Y = W2 @ H + Ws @ X + biasA[b][m] (+ biasB[m])
// ---------------------------------------------------------------------------
__global__ __launch_bounds__(128) void mlp_stage2_wmma_kernel(
    const float* __restrict__ W2, int ldw2,
    const float* __restrict__ Hm, int hBS, int Hk,
    const float* __restrict__ Ws, int ldws,
    const float* __restrict__ X, int xBS, int K,
    float* __restrict__ Y, int yBS,
    const float* __restrict__ biasA, int biasABS,
    const float* __restrict__ biasB, int M) {
  int tilesPerBatch = (M >> 4) << 4;
  int tile = blockIdx.x * 4 + (threadIdx.x >> 5);
  if (tile >= 64 * tilesPerBatch) return;
  int b  = tile / tilesPerBatch;
  int t  = tile - b * tilesPerBatch;
  int m0 = (t >> 4) << 4;
  int n0 = (t & 15) << 4;
  int lane = threadIdx.x & 31, half = lane >> 4, mr = lane & 15;
  const float* Hb = Hm + b * hBS;
  const float* Xb = X + b * xBS;
  v8f acc = {};
  for (int k0 = 0; k0 < Hk; k0 += 32) {         // W2 @ H
    v16h av, bv;
#pragma unroll
    for (int e = 0; e < 16; ++e) {
      int kk = k0 + kmap(e, half);
      av[e] = (_Float16)W2[(m0 + mr) * ldw2 + kk];
      bv[e] = (_Float16)Hb[kk * 256 + n0 + mr];
    }
    acc = __builtin_amdgcn_wmma_f32_16x16x32_f16(false, av, false, bv,
                                                 (short)0, acc, false, false);
  }
  for (int k0 = 0; k0 < K; k0 += 32) {          // + Ws @ X (shortcut)
    v16h av, bv;
#pragma unroll
    for (int e = 0; e < 16; ++e) {
      int kk = k0 + kmap(e, half);
      av[e] = (_Float16)Ws[(m0 + mr) * ldws + kk];
      bv[e] = (_Float16)Xb[kk * 256 + n0 + mr];
    }
    acc = __builtin_amdgcn_wmma_f32_16x16x32_f16(false, av, false, bv,
                                                 (short)0, acc, false, false);
  }
  int n = n0 + mr;
#pragma unroll
  for (int r = 0; r < 8; ++r) {
    int m = m0 + r + (half << 3);
    float v = acc[r] + biasA[b * biasABS + m];
    if (biasB) v += biasB[m];
    Y[b * yBS + m * 256 + n] = v;
  }
}

// ---------------------------------------------------------------------------
// Kernel 5: seed head:  seed[b,d,n] = sum_j m4_w2[d,j]*h4[b,j,n] + m4_b2[d]
// also scatters seed points into all_pts[b][n][d] (first 256 points).
// ---------------------------------------------------------------------------
__global__ void seed_kernel(const float* __restrict__ h4,
                            const float* __restrict__ w2,
                            const float* __restrict__ b2,
                            float* __restrict__ seed_out,
                            float* __restrict__ allpts) {
  int t = blockIdx.x * 256 + threadIdx.x;
  if (t >= 64 * 256) return;
  int b = t >> 8, n = t & 255;
  const float* H = h4 + b * 16384;              // [64,256] per batch
#pragma unroll
  for (int d = 0; d < 3; ++d) {
    float acc = b2[d];
    for (int j = 0; j < 64; ++j) acc += w2[d * 64 + j] * H[j * 256 + n];
    seed_out[b * 768 + d * 256 + n] = acc;      // [B,3,256]
    allpts[b * 13056 + n * 3 + d] = acc;        // [B,4352,3]
  }
}

// Kernel 6: append partial points: all_pts[b][256+j][:] = partial[b][j][:]
__global__ void copy_partial_kernel(const float* __restrict__ partial,
                                    float* __restrict__ allpts) {
  int t = blockIdx.x * 256 + threadIdx.x;
  if (t >= 64 * 4096 * 3) return;
  int b = t / 12288;
  int r = t - b * 12288;
  allpts[b * 13056 + 768 + r] = partial[t];
}

// ---------------------------------------------------------------------------
// Kernel 7: farthest point sampling, one workgroup per batch.
// dist[] stays resident in LDS across the 512 serial steps; argmax matches
// jnp.argmax (first occurrence) via strict-> local scan + lower-index ties.
// ---------------------------------------------------------------------------
__global__ __launch_bounds__(256) void fps_kernel(const float* __restrict__ pts,
                                                  float* __restrict__ xout) {
  const int N = 4352;
  int b = blockIdx.x;
  const float* P = pts + b * N * 3;
  __shared__ float dist[4352];
  __shared__ int   sel[512];
  __shared__ float rv[256];
  __shared__ int   ri[256];
  __shared__ int   farSh;
  int tid = threadIdx.x;
  for (int i = tid; i < N; i += 256) dist[i] = 1e10f;
  if (tid == 0) farSh = 0;
  __syncthreads();

  for (int t = 0; t < 512; ++t) {
    int far = farSh;
    if (tid == 0) sel[t] = far;                 // record BEFORE update (scan ys)
    float cx = P[far * 3 + 0], cy = P[far * 3 + 1], cz = P[far * 3 + 2];
    float bvv = -1.f;
    int   bii = 0x7fffffff;
    for (int i = tid; i < N; i += 256) {
      float dx = P[i * 3 + 0] - cx;
      float dy = P[i * 3 + 1] - cy;
      float dz = P[i * 3 + 2] - cz;
      float d  = dx * dx + dy * dy + dz * dz;
      float nd = fminf(dist[i], d);
      dist[i] = nd;
      if (nd > bvv) { bvv = nd; bii = i; }      // strict: first max per thread
    }
    rv[tid] = bvv; ri[tid] = bii;
    __syncthreads();
    for (int s = 128; s > 0; s >>= 1) {
      if (tid < s) {
        float ov = rv[tid + s]; int oi = ri[tid + s];
        if (ov > rv[tid] || (ov == rv[tid] && oi < ri[tid])) {
          rv[tid] = ov; ri[tid] = oi;
        }
      }
      __syncthreads();
    }
    if (tid == 0) farSh = ri[0];
    __syncthreads();
  }
  // gather to output [B,3,512]
  for (int t = tid; t < 512; t += 256) {
    int idx = sel[t];
    xout[b * 1536 +    0 + t] = P[idx * 3 + 0];
    xout[b * 1536 +  512 + t] = P[idx * 3 + 1];
    xout[b * 1536 + 1024 + t] = P[idx * 3 + 2];
  }
}

// ---------------------------------------------------------------------------
extern "C" void kernel_launch(void* const* d_in, const int* in_sizes, int n_in,
                              void* d_out, int out_size, void* d_ws, size_t ws_size,
                              hipStream_t stream) {
  const float* feat    = (const float*)d_in[0];
  // d_in[1] patch_xyz, d_in[2] patch_feat: unused by the reference
  const float* partial = (const float*)d_in[3];
  const float* ps_w    = (const float*)d_in[4];
  const float* ps_b    = (const float*)d_in[5];
  const float* m1_w1 = (const float*)d_in[6];   const float* m1_b1 = (const float*)d_in[7];
  const float* m1_w2 = (const float*)d_in[8];   const float* m1_b2 = (const float*)d_in[9];
  const float* m1_ws = (const float*)d_in[10];  const float* m1_bs = (const float*)d_in[11];
  const float* m2_w1 = (const float*)d_in[12];  const float* m2_b1 = (const float*)d_in[13];
  const float* m2_w2 = (const float*)d_in[14];  const float* m2_b2 = (const float*)d_in[15];
  const float* m2_ws = (const float*)d_in[16];  const float* m2_bs = (const float*)d_in[17];
  const float* m3_w1 = (const float*)d_in[18];  const float* m3_b1 = (const float*)d_in[19];
  const float* m3_w2 = (const float*)d_in[20];  const float* m3_b2 = (const float*)d_in[21];
  const float* m3_ws = (const float*)d_in[22];  const float* m3_bs = (const float*)d_in[23];
  const float* m4_w1 = (const float*)d_in[24];  const float* m4_b1 = (const float*)d_in[25];
  const float* m4_w2 = (const float*)d_in[26];  const float* m4_b2 = (const float*)d_in[27];

  float* out      = (float*)d_out;
  float* seed_out = out;                        // [64,3,256]   = 49152
  float* x3_out   = out + 49152;                // [64,128,256] = 2097152
  float* x_out    = out + 49152 + 2097152;      // [64,3,512]   = 98304

  float* ws     = (float*)d_ws;
  float* bufA   = ws;                           // [64,128,256]
  float* bufB   = ws + 2097152;                 // [64,128,256]
  float* hbuf   = ws + 4194304;                 // [64,128,256] (also m4 hidden)
  float* allpts = ws + 6291456;                 // [64,4352,3]
  float* fb1    = ws + 7127040;                 // 4 x [64,128]
  float* fbs1   = fb1 + 8192;
  float* fb3    = fbs1 + 8192;
  float* fbs3   = fb3 + 8192;

  // feat-projection bias vectors (folds concat's broadcast half + biases)
  feat_bias_kernel<<<32, 256, 0, stream>>>(feat, m1_w1, m1_b1, m1_b2, m1_ws,
                                           m1_bs, m3_w1, m3_b1, m3_b2, m3_ws,
                                           m3_bs, fb1, fbs1, fb3, fbs3);
  // x1 = deconv(feat) + ps_b  -> bufA
  deconv_wmma_kernel<<<2048, 128, 0, stream>>>(feat, ps_w, ps_b, bufA);
  // MLP_Res 1: in 640ch (x1 + feat) -> 128ch  (bufA -> bufB)
  mlp_stage1_wmma_kernel<<<2048, 128, 0, stream>>>(m1_w1, 640, bufA, 32768,
                                                   hbuf, 32768, fb1, 128,
                                                   128, 128, 1);
  mlp_stage2_wmma_kernel<<<2048, 128, 0, stream>>>(m1_w2, 128, hbuf, 32768, 128,
                                                   m1_ws, 640, bufA, 32768, 128,
                                                   bufB, 32768, fbs1, 128,
                                                   nullptr, 128);
  // MLP_Res 2: 128 -> 64 -> 128  (bufB -> bufA)
  mlp_stage1_wmma_kernel<<<1024, 128, 0, stream>>>(m2_w1, 128, bufB, 32768,
                                                   hbuf, 16384, m2_b1, 0,
                                                   64, 128, 1);
  mlp_stage2_wmma_kernel<<<2048, 128, 0, stream>>>(m2_w2, 64, hbuf, 16384, 64,
                                                   m2_ws, 128, bufB, 32768, 128,
                                                   bufA, 32768, m2_bs, 0,
                                                   m2_b2, 128);
  // MLP_Res 3: in 640ch (x2 + feat) -> 128ch  (bufA -> x3_out)
  mlp_stage1_wmma_kernel<<<2048, 128, 0, stream>>>(m3_w1, 640, bufA, 32768,
                                                   hbuf, 32768, fb3, 128,
                                                   128, 128, 1);
  mlp_stage2_wmma_kernel<<<2048, 128, 0, stream>>>(m3_w2, 128, hbuf, 32768, 128,
                                                   m3_ws, 640, bufA, 32768, 128,
                                                   x3_out, 32768, fbs3, 128,
                                                   nullptr, 128);
  // m4 hidden: relu(m4_w1 @ x3 + b1) -> hbuf (64ch)
  mlp_stage1_wmma_kernel<<<1024, 128, 0, stream>>>(m4_w1, 128, x3_out, 32768,
                                                   hbuf, 16384, m4_b1, 0,
                                                   64, 128, 1);
  // seed = m4_w2 @ hidden + b2; also build all_pts
  seed_kernel<<<64, 256, 0, stream>>>(hbuf, m4_w2, m4_b2, seed_out, allpts);
  copy_partial_kernel<<<3072, 256, 0, stream>>>(partial, allpts);
  // FPS(512) -> x_out [64,3,512]
  fps_kernel<<<64, 256, 0, stream>>>(allpts, x_out);
}
